// Generator_16638703305114
// MI455X (gfx1250) — matrix operand order repaired and checked by reading, not compiled, not run on hardware
//
#include <hip/hip_runtime.h>
#include <math.h>

typedef __attribute__((ext_vector_type(16))) _Float16 v16h;
typedef __attribute__((ext_vector_type(8)))  _Float16 v8h;
typedef __attribute__((ext_vector_type(8)))  float    v8f;

template <bool RA, bool RB>
__device__ __forceinline__ v8f wmma16(v16h a, v16h b, v8f c) {
    return __builtin_amdgcn_wmma_f32_16x16x32_f16(false, a, false, b,
                                                  (short)0, c, RA, RB);
}

// ---------------------------------------------------------------------------
// NT GEMM, f16 in / f16 out: C[M,N] = scale*(A[M,K] x B[N,K]^T) + bias, ReLU opt.
// Block tile 128x128x32, 256 threads = 8 waves; wave = 32x64 = 8 WMMA tiles.
// Double-buffered LDS fed by GLOBAL_LOAD_ASYNC_TO_LDS_B128 (ASYNCcnt pipeline).
// B is staged directly in per-lane WMMA fragment layout (each lane fragment is
// 32 contiguous bytes of B in global memory -> pure async b128 copies).
// Compute: preload 2 A-frags + 4 B-frags, then 8 WMMAs snake-ordered with
// ISA 7.12 operand-reuse hints (RA/RB) as compile-time constants.
// ---------------------------------------------------------------------------
__global__ __launch_bounds__(256)
void gemm_nt_f16_kernel(const _Float16* __restrict__ A,
                        const _Float16* __restrict__ B,
                        const float* __restrict__ bias,
                        _Float16* __restrict__ C,
                        int M, int N, int K,
                        long long sA, long long sB, long long sC,
                        float scale, int relu) {
    constexpr int BM = 128, BN = 128, BK = 32, PA = 8;
    __shared__ _Float16 As[2][BM][BK + PA];   // K-contiguous rows
    __shared__ v16h     Bf[2][BN / 16][32];   // [n-tile][lane] = B fragment

    const long long zb = blockIdx.z;
    A += zb * sA; B += zb * sB; C += zb * sC;

    const int tid  = threadIdx.x;
    const int lane = tid & 31;
    const int wid  = tid >> 5;
    const int wm   = wid >> 1;            // 0..3 -> M offset wm*32
    const int wn   = wid & 1;             // 0..1 -> N offset wn*64
    const int gm0  = blockIdx.y * BM;
    const int gn0  = blockIdx.x * BN;

    // staging coordinates (per thread: 32B of A, 32B of B per stage)
    const int ar  = tid >> 1;             // 0..127 A row
    const int acb = (tid & 1) * 16;       // 0,16   A k-offset (halves)
    const int bn  = tid >> 1;             // 0..127 B row (n)
    const int bkh = tid & 1;              // k-group
    const _Float16* gA = A + (long long)(gm0 + ar) * K + acb;
    const _Float16* gB = B + (long long)(gn0 + bn) * K + bkh * 16;

    auto issue = [&](int buf, int kk) {
        unsigned la = (unsigned)(unsigned long long)(const void*)&As[buf][ar][acb];
        unsigned lb = (unsigned)(unsigned long long)
                          (const void*)&Bf[buf][bn >> 4][bkh * 16 + (bn & 15)];
        unsigned long long ga = (unsigned long long)(const void*)(gA + kk);
        unsigned long long gb = (unsigned long long)(const void*)(gB + kk);
        asm volatile("global_load_async_to_lds_b128 %0, %1, off\n\t"
                     "global_load_async_to_lds_b128 %0, %1, off offset:16"
                     :: "v"(la), "v"(ga) : "memory");
        asm volatile("global_load_async_to_lds_b128 %0, %1, off\n\t"
                     "global_load_async_to_lds_b128 %0, %1, off offset:16"
                     :: "v"(lb), "v"(gb) : "memory");
    };

    v8f acc[2][4] = {};
    const int nk = K / BK;

    issue(0, 0);
    for (int kt = 0; kt < nk; ++kt) {
        const int cur = kt & 1;
        if (kt + 1 < nk) {
            issue(cur ^ 1, (kt + 1) * BK);
            asm volatile("s_wait_asynccnt 0x4" ::: "memory");
        } else {
            asm volatile("s_wait_asynccnt 0x0" ::: "memory");
        }
        __syncthreads();   // stage `cur` fully resident for all waves

        // preload all fragments: one ds clause, then back-to-back WMMAs
        union { v16h v; v8h h[2]; } a0, a1;
        const int am  = wm * 32 + (lane & 15);
        const int akb = (lane >> 4) * 16;
        a0.h[0] = *(const v8h*)&As[cur][am][akb];
        a0.h[1] = *(const v8h*)&As[cur][am][akb + 8];
        a1.h[0] = *(const v8h*)&As[cur][am + 16][akb];
        a1.h[1] = *(const v8h*)&As[cur][am + 16][akb + 8];
        v16h bf0 = Bf[cur][wn * 4 + 0][lane];
        v16h bf1 = Bf[cur][wn * 4 + 1][lane];
        v16h bf2 = Bf[cur][wn * 4 + 2][lane];
        v16h bf3 = Bf[cur][wn * 4 + 3][lane];

        // snake order: maximize A/B operand reuse between adjacent WMMAs
        acc[0][0] = wmma16<false, false>(a0.v, bf0, acc[0][0]);
        acc[0][1] = wmma16<true,  false>(a0.v, bf1, acc[0][1]);  // reuse A
        acc[0][2] = wmma16<true,  false>(a0.v, bf2, acc[0][2]);  // reuse A
        acc[0][3] = wmma16<true,  false>(a0.v, bf3, acc[0][3]);  // reuse A
        acc[1][3] = wmma16<false, true >(a1.v, bf3, acc[1][3]);  // reuse B
        acc[1][2] = wmma16<true,  false>(a1.v, bf2, acc[1][2]);  // reuse A
        acc[1][1] = wmma16<true,  false>(a1.v, bf1, acc[1][1]);  // reuse A
        acc[1][0] = wmma16<true,  false>(a1.v, bf0, acc[1][0]);  // reuse A

        __syncthreads();   // all waves done with `cur` before it is refilled
    }

    // epilogue: C/D layout (lane: col = lane&15, rows (lane>>4)*8 + v)
    const int col   = lane & 15;
    const int rbase = (lane >> 4) << 3;
    #pragma unroll
    for (int i = 0; i < 2; ++i) {
        #pragma unroll
        for (int t = 0; t < 4; ++t) {
            int c = gn0 + wn * 64 + t * 16 + col;
            float bv = bias ? bias[c] : 0.0f;
            #pragma unroll
            for (int v = 0; v < 8; ++v) {
                int r = gm0 + wm * 32 + i * 16 + rbase + v;
                float val = acc[i][t][v] * scale + bv;
                if (relu) val = fmaxf(val, 0.0f);
                C[(long long)r * N + c] = (_Float16)val;
            }
        }
    }
}

// ---------------------------------------------------------------------------
// f32 -> f16 straight conversion, 8 elems/thread
// ---------------------------------------------------------------------------
__global__ __launch_bounds__(256)
void cvt_f32_f16_kernel(const float* __restrict__ in, _Float16* __restrict__ out,
                        long long n) {
    long long i = ((long long)blockIdx.x * 256 + threadIdx.x) * 8;
    if (i + 8 > n) return;
    float4 f0 = ((const float4*)(in + i))[0];
    float4 f1 = ((const float4*)(in + i))[1];
    v8h t;
    t[0] = (_Float16)f0.x; t[1] = (_Float16)f0.y;
    t[2] = (_Float16)f0.z; t[3] = (_Float16)f0.w;
    t[4] = (_Float16)f1.x; t[5] = (_Float16)f1.y;
    t[6] = (_Float16)f1.z; t[7] = (_Float16)f1.w;
    *(v8h*)(out + i) = t;
}

// ---------------------------------------------------------------------------
// Batched tiled transpose: out[c][r] = (f16) in[r][c]   (32x32 tiles via LDS)
// ---------------------------------------------------------------------------
template <typename T>
__global__ __launch_bounds__(256)
void tr_to_f16_kernel(const T* __restrict__ in, _Float16* __restrict__ out,
                      int R, int C, long long sIn, long long sOut) {
    __shared__ _Float16 tile[32][33];
    in  += (long long)blockIdx.z * sIn;
    out += (long long)blockIdx.z * sOut;
    const int tx = threadIdx.x, ty = threadIdx.y;     // 32 x 8
    const int c0 = blockIdx.x * 32, r0 = blockIdx.y * 32;
    #pragma unroll
    for (int j = 0; j < 4; ++j)
        tile[ty + j * 8][tx] = (_Float16)in[(long long)(r0 + ty + j * 8) * C + c0 + tx];
    __syncthreads();
    #pragma unroll
    for (int j = 0; j < 4; ++j)
        out[(long long)(c0 + ty + j * 8) * R + r0 + tx] = tile[tx][ty + j * 8];
}

// ---------------------------------------------------------------------------
__device__ __forceinline__ float blk_sum(float v, float* red, int n) {
    int tid = threadIdx.x;
    red[tid] = v; __syncthreads();
    for (int s = n >> 1; s > 0; s >>= 1) {
        if (tid < s) red[tid] += red[tid + s];
        __syncthreads();
    }
    float r = red[0]; __syncthreads();
    return r;
}
__device__ __forceinline__ float blk_max(float v, float* red, int n) {
    int tid = threadIdx.x;
    red[tid] = v; __syncthreads();
    for (int s = n >> 1; s > 0; s >>= 1) {
        if (tid < s) red[tid] = fmaxf(red[tid], red[tid + s]);
        __syncthreads();
    }
    float r = red[0]; __syncthreads();
    return r;
}

// blended = tt*softmax(mixed) + (1-tt)*softmax(att), in-place into mixed (f16)
__global__ __launch_bounds__(256)
void softmax_blend_kernel(_Float16* __restrict__ mixed,
                          const _Float16* __restrict__ att, int T1, float tt) {
    __shared__ float red[256];
    long long r = blockIdx.x;
    _Float16* mrow = mixed + r * (long long)T1;
    const _Float16* arow = att + r * (long long)T1;
    int tid = threadIdx.x;

    float m1 = -3.4e38f, m2 = -3.4e38f;
    for (int i = tid; i < T1; i += 256) {
        m1 = fmaxf(m1, (float)mrow[i]);
        m2 = fmaxf(m2, (float)arow[i]);
    }
    m1 = blk_max(m1, red, 256);
    m2 = blk_max(m2, red, 256);

    float s1 = 0.f, s2 = 0.f;
    for (int i = tid; i < T1; i += 256) {
        s1 += __expf((float)mrow[i] - m1);
        s2 += __expf((float)arow[i] - m2);
    }
    s1 = blk_sum(s1, red, 256);
    s2 = blk_sum(s2, red, 256);

    float i1 = tt / s1, i2 = (1.0f - tt) / s2;
    for (int i = tid; i < T1; i += 256)
        mrow[i] = (_Float16)(__expf((float)mrow[i] - m1) * i1 +
                             __expf((float)arow[i] - m2) * i2);
}

// x = LayerNorm(x + res)*g + b ; writes f32 x and f16 shadow xh. D=384=3*128.
__global__ __launch_bounds__(128)
void add_ln_kernel(float* __restrict__ x, _Float16* __restrict__ xh,
                   const _Float16* __restrict__ res,
                   const float* __restrict__ g, const float* __restrict__ be,
                   int D) {
    __shared__ float red[128];
    long long r = blockIdx.x;
    float* xr = x + r * (long long)D;
    _Float16* xhr = xh + r * (long long)D;
    const _Float16* rr = res + r * (long long)D;
    int tid = threadIdx.x;

    float loc[3]; float s = 0.f;
    #pragma unroll
    for (int j = 0; j < 3; ++j) {
        int i = tid + j * 128;
        loc[j] = xr[i] + (float)rr[i];
        s += loc[j];
    }
    float mean = blk_sum(s, red, 128) / (float)D;
    float vs = 0.f;
    #pragma unroll
    for (int j = 0; j < 3; ++j) { float d = loc[j] - mean; vs += d * d; }
    float var = blk_sum(vs, red, 128) / (float)D;
    float inv = rsqrtf(var + 1e-5f);
    #pragma unroll
    for (int j = 0; j < 3; ++j) {
        int i = tid + j * 128;
        float o = (loc[j] - mean) * inv * g[i] + be[i];
        xr[i] = o;
        xhr[i] = (_Float16)o;
    }
}

// ---------------------------------------------------------------------------
static inline void launch_gemm(hipStream_t stream, const _Float16* A,
                               const _Float16* B, const float* bias, _Float16* C,
                               int M, int N, int K,
                               long long sA, long long sB, long long sC, int batch,
                               float scale, int relu) {
    dim3 grid(N / 128, M / 128, batch);
    gemm_nt_f16_kernel<<<grid, dim3(256), 0, stream>>>(A, B, bias, C, M, N, K,
                                                       sA, sB, sC, scale, relu);
}
static inline void launch_cvt(hipStream_t stream, const float* in, _Float16* out,
                              long long n) {
    cvt_f32_f16_kernel<<<dim3((unsigned)(n / 2048)), dim3(256), 0, stream>>>(in, out, n);
}

extern "C" void kernel_launch(void* const* d_in, const int* in_sizes, int n_in,
                              void* d_out, int out_size, void* d_ws, size_t ws_size,
                              hipStream_t stream) {
    const int kL = 4, kB = 4, kT = 1024, kTC = 1024, kT1 = 2048,
              kD = 384, kDK = 256, kFF = 1536;
    const float kTT = 0.1f;

    const float* tgt     = (const float*)d_in[0];
    const float* memory  = (const float*)d_in[1];
    const float* score_c = (const float*)d_in[2];
    const float* out_c   = (const float*)d_in[3];
    const float* Wq  = (const float*)d_in[4];
    const float* bq  = (const float*)d_in[5];
    const float* Wk  = (const float*)d_in[6];
    const float* bk  = (const float*)d_in[7];
    const float* Wv  = (const float*)d_in[8];
    const float* bv  = (const float*)d_in[9];
    const float* Wkn = (const float*)d_in[10];
    const float* bkn = (const float*)d_in[11];
    const float* Wun = (const float*)d_in[12];
    const float* bun = (const float*)d_in[13];
    const float* W1  = (const float*)d_in[14];
    const float* b1  = (const float*)d_in[15];
    const float* W2  = (const float*)d_in[16];
    const float* b2  = (const float*)d_in[17];
    const float* g1  = (const float*)d_in[18];
    const float* be1 = (const float*)d_in[19];
    const float* g2  = (const float*)d_in[20];
    const float* be2 = (const float*)d_in[21];
    (void)in_sizes; (void)n_in; (void)out_size; (void)ws_size;

    // ---- workspace carve (byte-based, all chunks 16B aligned) ----
    char* wsb = (char*)d_ws;
    auto allocF = [&](size_t n) { float* p = (float*)wsb; wsb += n * 4; return p; };
    auto allocH = [&](size_t n) { _Float16* p = (_Float16*)wsb; wsb += n * 2; return p; };

    const size_t nBT   = (size_t)kB * kT  * kD;
    const size_t nMem  = (size_t)kB * kT1 * kDK;
    const size_t nKV   = (size_t)kB * kT1 * kD;
    const size_t nAtt  = (size_t)kB * kT  * kT1;
    const size_t nKno  = (size_t)kB * kTC * kD;
    const size_t nS1   = (size_t)kB * kT  * kTC;
    const size_t nH    = (size_t)kB * kT  * kFF;
    const size_t nSc   = (size_t)kB * kTC * kT1;

    float*    xb   = allocF(nBT);     // running x (f32, LN precision)
    _Float16* xh   = allocH(nBT);     // f16 shadow of x
    _Float16* qh   = allocH(nBT);     // q / unk / y2
    _Float16* kh   = allocH(nKV);
    _Float16* vh   = allocH(nKV);
    _Float16* vT   = allocH(nKV);     // v transposed [b][D][T1]
    _Float16* atth = allocH(nAtt);
    _Float16* knoh = allocH(nKno);    // kno / z
    _Float16* s1h  = allocH(nS1);
    _Float16* mixh = allocH(nAtt);    // mixed / blended
    _Float16* hh   = allocH(nH);
    _Float16* memh = allocH(nMem);
    _Float16* ocH  = allocH(nKno);    // out_c[l] f16 (per layer)
    _Float16* scT  = allocH(nSc);     // score_c[l]^T f16 (per layer)
    _Float16* WqH  = allocH((size_t)kL * kD * kD);
    _Float16* WkH  = allocH((size_t)kL * kD * kDK);
    _Float16* WvH  = allocH((size_t)kL * kD * kDK);
    _Float16* WknH = allocH((size_t)kL * kD * kD);
    _Float16* WunH = allocH((size_t)kL * kD * kD);
    _Float16* W1H  = allocH((size_t)kL * kFF * kD);
    _Float16* W2H  = allocH((size_t)kL * kD * kFF);

    const float iscale = 1.0f / sqrtf((float)kD);
    const size_t xbytes = sizeof(float) * nBT;

    // ---- one-time f32 -> f16 conversions ----
    (void)hipMemcpyAsync(xb, tgt, xbytes, hipMemcpyDeviceToDevice, stream);
    launch_cvt(stream, tgt, xh, (long long)nBT);
    launch_cvt(stream, memory, memh, (long long)nMem);
    launch_cvt(stream, Wq,  WqH,  (long long)kL * kD * kD);
    launch_cvt(stream, Wk,  WkH,  (long long)kL * kD * kDK);
    launch_cvt(stream, Wv,  WvH,  (long long)kL * kD * kDK);
    launch_cvt(stream, Wkn, WknH, (long long)kL * kD * kD);
    launch_cvt(stream, Wun, WunH, (long long)kL * kD * kD);
    launch_cvt(stream, W1,  W1H,  (long long)kL * kFF * kD);
    launch_cvt(stream, W2,  W2H,  (long long)kL * kD * kFF);

    for (int l = 0; l < kL; ++l) {
        const _Float16* Wq_l  = WqH  + (size_t)l * kD * kD;
        const _Float16* Wk_l  = WkH  + (size_t)l * kD * kDK;
        const _Float16* Wv_l  = WvH  + (size_t)l * kD * kDK;
        const _Float16* Wkn_l = WknH + (size_t)l * kD * kD;
        const _Float16* Wun_l = WunH + (size_t)l * kD * kD;
        const _Float16* W1_l  = W1H  + (size_t)l * kFF * kD;
        const _Float16* W2_l  = W2H  + (size_t)l * kD * kFF;

        // per-layer operand prep: out_c[l] -> f16 ; score_c[l] -> f16 transposed
        launch_cvt(stream, out_c + (size_t)l * nKno, ocH, (long long)nKno);
        tr_to_f16_kernel<float><<<dim3(kT1 / 32, kTC / 32, kB), dim3(32, 8), 0, stream>>>(
            score_c + (size_t)l * nSc, scT, kTC, kT1,
            (long long)kTC * kT1, (long long)kT1 * kTC);

        // q = x @ Wq^T + bq
        launch_gemm(stream, xh, Wq_l, bq + l * kD, qh,
                    kB * kT, kD, kD, 0, 0, 0, 1, 1.0f, 0);
        // k = memory @ Wk^T + bk ; v = memory @ Wv^T + bv
        launch_gemm(stream, memh, Wk_l, bk + l * kD, kh,
                    kB * kT1, kD, kDK, 0, 0, 0, 1, 1.0f, 0);
        launch_gemm(stream, memh, Wv_l, bv + l * kD, vh,
                    kB * kT1, kD, kDK, 0, 0, 0, 1, 1.0f, 0);
        // v^T per batch: [T1,D] -> [D,T1]
        tr_to_f16_kernel<_Float16><<<dim3(kD / 32, kT1 / 32, kB), dim3(32, 8), 0, stream>>>(
            vh, vT, kT1, kD, (long long)kT1 * kD, (long long)kD * kT1);
        // att = (q @ k^T) * scale   per batch
        launch_gemm(stream, qh, kh, nullptr, atth,
                    kT, kT1, kD,
                    (long long)kT * kD, (long long)kT1 * kD, (long long)kT * kT1,
                    kB, iscale, 0);
        // unk = x @ Wun^T + bun   (reuses qh)
        launch_gemm(stream, xh, Wun_l, bun + l * kD, qh,
                    kB * kT, kD, kD, 0, 0, 0, 1, 1.0f, 0);
        // kno = out_c[l] @ Wkn^T + bkn
        launch_gemm(stream, ocH, Wkn_l, bkn + l * kD, knoh,
                    kB * kTC, kD, kD, 0, 0, 0, 1, 1.0f, 0);
        // score1 = (unk @ kno^T) * scale   per batch
        launch_gemm(stream, qh, knoh, nullptr, s1h,
                    kT, kTC, kD,
                    (long long)kT * kD, (long long)kTC * kD, (long long)kT * kTC,
                    kB, iscale, 0);
        // mixed = score1 @ score_c[l] = score1 @ (scT)^T   per batch (NT)
        launch_gemm(stream, s1h, scT, nullptr, mixh,
                    kT, kT1, kTC,
                    (long long)kT * kTC, (long long)kT1 * kTC, (long long)kT * kT1,
                    kB, 1.0f, 0);
        // blended (in-place in mixh)
        softmax_blend_kernel<<<dim3(kB * kT), dim3(256), 0, stream>>>(
            mixh, atth, kT1, kTT);
        // z = blended @ v = blended @ (vT)^T   per batch (NT), into knoh
        launch_gemm(stream, mixh, vT, nullptr, knoh,
                    kT, kD, kT1,
                    (long long)kT * kT1, (long long)kD * kT1, (long long)kT * kD,
                    kB, 1.0f, 0);
        // x = LN(x + z)
        add_ln_kernel<<<dim3(kB * kT), dim3(128), 0, stream>>>(
            xb, xh, knoh, g1 + l * kD, be1 + l * kD, kD);
        // h = relu(x @ W1^T + b1)
        launch_gemm(stream, xh, W1_l, b1 + l * kFF, hh,
                    kB * kT, kFF, kD, 0, 0, 0, 1, 1.0f, 1);
        // y2 = h @ W2^T + b2  (into qh)
        launch_gemm(stream, hh, W2_l, b2 + l * kD, qh,
                    kB * kT, kD, kFF, 0, 0, 0, 1, 1.0f, 0);
        // x = LN(x + y2)
        add_ln_kernel<<<dim3(kB * kT), dim3(128), 0, stream>>>(
            xb, xh, qh, g2 + l * kD, be2 + l * kD, kD);
    }

    (void)hipMemcpyAsync(d_out, xb, xbytes, hipMemcpyDeviceToDevice, stream);
}
